// PET_36481452212975
// MI455X (gfx1250) — matrix operand order repaired and checked by reading, not compiled, not run on hardware
//
#include <hip/hip_runtime.h>
#include <hip/hip_bf16.h>

// ---------------- problem constants ----------------
#define Dd   64
#define D2d  128
#define Nn   20000
#define Rr   24
#define Ss   3
#define Ee   60000
#define Bb   8
#define Kk   101
#define HLn  50
#define Mrows (Bb * Nn)   // 160000

typedef __attribute__((ext_vector_type(16))) _Float16 v16h;
typedef __attribute__((ext_vector_type(8)))  float    v8f;
typedef __attribute__((ext_vector_type(4)))  int      v4i;

// ---------------- CDNA5 async global->LDS staging helpers ----------------
#if __has_builtin(__builtin_amdgcn_global_load_async_to_lds_b128)
#define PET_HAS_ASYNC 1
#endif

__device__ __forceinline__ void pet_copy_b128_to_lds(const float* __restrict__ g,
                                                     float* __restrict__ lds) {
#ifdef PET_HAS_ASYNC
    __builtin_amdgcn_global_load_async_to_lds_b128(
        (__attribute__((address_space(1))) v4i*)g,
        (__attribute__((address_space(3))) v4i*)lds,
        0, 0);
#else
    *(float4*)lds = *(const float4*)g;
#endif
}

__device__ __forceinline__ void pet_async_wait() {
#ifdef PET_HAS_ASYNC
#if __has_builtin(__builtin_amdgcn_s_wait_asynccnt)
    __builtin_amdgcn_s_wait_asynccnt(0);
#else
    asm volatile("s_wait_asynccnt 0x0" ::: "memory");
#endif
#endif
}

// ---------------- utility kernels ----------------

__global__ void pet_zero_kernel(float4* __restrict__ p, long n4) {
    long id = (long)blockIdx.x * blockDim.x + threadIdx.x;
    if (id < n4) { float4 z; z.x = 0.f; z.y = 0.f; z.z = 0.f; z.w = 0.f; p[id] = z; }
}

// degree histogram (segment_sum of poi_emb over its own indices == deg * emb)
__global__ void pet_degree_kernel(const int* __restrict__ src, const int* __restrict__ dst,
                                  float* __restrict__ deg) {
    int id = blockIdx.x * blockDim.x + threadIdx.x;
    if (id < Ee)              unsafeAtomicAdd(deg + src[id], 1.0f);
    else if (id < 2 * Ee)     unsafeAtomicAdd(deg + dst[id - Ee], 1.0f);
}

// Weight staging: f32 -> f16, pre-swizzled into WMMA B-fragment order so each
// lane's 16 halfs are contiguous (32B): frag index (kc*4+ct), lane = (kr>=16)*16 + n%16,
// element e = kr&15, where k = kc*32+kr.
__global__ void pet_prep_kernel(const float* __restrict__ gate_W, const float* __restrict__ gnn_W,
                                const float* __restrict__ query_emb, const int* __restrict__ r_index,
                                _Float16* __restrict__ wswz_gate, _Float16* __restrict__ wswz_gnn,
                                float* __restrict__ query) {
    int id = blockIdx.x * blockDim.x + threadIdx.x;
    if (id < 2 * 128 * 64) {                     // gnn_W [2][128][64]
        int l = id >> 13, rem = id & 8191;
        int k = rem >> 6, n = rem & 63;
        int kc = k >> 5, kr = k & 31;
        int lane = ((kr >> 4) << 4) + (n & 15);
        wswz_gnn[(long)l * 8192 + (((kc * 4 + (n >> 4)) * 32 + lane) << 4) + (kr & 15)] =
            (_Float16)gnn_W[id];
    }
    if (id < 64 * 64) {                          // gate_W [64][64]
        int k = id >> 6, n = id & 63;
        int kc = k >> 5, kr = k & 31;
        int lane = ((kr >> 4) << 4) + (n & 15);
        wswz_gate[(((kc * 4 + (n >> 4)) * 32 + lane) << 4) + (kr & 15)] = (_Float16)gate_W[id];
    }
    if (id < Bb * Dd) {
        int b = id >> 6;
        query[id] = query_emb[r_index[b] * Dd + (id & 63)];
    }
}

// snapshot 0: x = init = deg[n] * poi_emb[n,:]
__global__ void pet_init0_kernel(const float* __restrict__ poi, const float* __restrict__ deg,
                                 float* __restrict__ x, float* __restrict__ initb) {
    long id = (long)blockIdx.x * blockDim.x + threadIdx.x;   // B*N*16 float4s
    if (id >= (long)Mrows * 16) return;
    int q = id & 15; long bn = id >> 4;
    int n = (int)(bn % Nn);
    float dv = deg[n];
    float4 v = ((const float4*)(poi + (long)n * Dd))[q];
    float4 o; o.x = v.x * dv; o.y = v.y * dv; o.z = v.z * dv; o.w = v.w * dv;
    ((float4*)(x + bn * Dd))[q]     = o;
    ((float4*)(initb + bn * Dd))[q] = o;
}

// edge message scatter: agg[b,dst,:] += x[b,src,:] + rel[et,:]
__global__ void pet_scatter_kernel(const int* __restrict__ src, const int* __restrict__ dst,
                                   const int* __restrict__ et, const float* __restrict__ x,
                                   const float* __restrict__ rel, float* __restrict__ agg) {
    long id = (long)blockIdx.x * blockDim.x + threadIdx.x;   // E*B*16
    if (id >= (long)Ee * Bb * 16) return;
    int q = id & 15; long t = id >> 4;
    int b = (int)(t & 7); long e = t >> 3;
    int sv = src[e], dv = dst[e], rv = et[e];
    float4 xs = ((const float4*)(x + ((long)b * Nn + sv) * Dd))[q];
    float4 rr = ((const float4*)(rel + (long)rv * Dd))[q];
    float* out = agg + ((long)b * Nn + dv) * Dd + q * 4;
    unsafeAtomicAdd(out + 0, xs.x + rr.x);
    unsafeAtomicAdd(out + 1, xs.y + rr.y);
    unsafeAtomicAdd(out + 2, xs.z + rr.z);
    unsafeAtomicAdd(out + 3, xs.w + rr.w);
}

// ---------------- WMMA GEMM kernels ----------------
// A fragment (16x32 f16): lane L holds row m=L&15; K run [2*kb..2*kb+15] (element e -> k=2*kb+e)
//   with kb=(L>>4)*8 -> one contiguous 16-float (64B) read per kc from the row-major f32 tile.
// B fragment: pre-swizzled in global memory, one 32B contiguous read per lane.
// C/D: VGPR r, lane L -> row r + 8*(L>>4), col L&15.

// x_out = relu([agg | init] @ W(128x64) + bias) ; 64 rows per block, 4 waves
__global__ __launch_bounds__(128) void pet_gnn_gemm_kernel(
        const float* __restrict__ agg, const float* __restrict__ initb,
        const _Float16* __restrict__ Wswz, const float* __restrict__ bias,
        float* __restrict__ xout) {
    __shared__ __align__(16) float Af[64][128];   // f32 A tile (cols 0-63 agg, 64-127 init)
    const int t = threadIdx.x;
    const int wave = t >> 5, lane = t & 31;
    const long rowBase = (long)blockIdx.x * 64;

    { // stage A tile via async DMA (16B per copy, 16 copies per thread)
        int r = t >> 1, half = t & 1;
        const float* srcp = (half == 0 ? agg : initb) + (rowBase + r) * Dd;
        #pragma unroll
        for (int c4 = 0; c4 < 16; ++c4)
            pet_copy_b128_to_lds(srcp + c4 * 4, &Af[r][half * 64 + c4 * 4]);
    }
    pet_async_wait();
    __syncthreads();

    const v16h* Wf = (const v16h*)Wswz;
    v8f acc[4] = {};
    const int m = lane & 15;
    const int kb = (lane >> 4) * 8;
    #pragma unroll
    for (int kc = 0; kc < 4; ++kc) {
        v16h a;
        #pragma unroll
        for (int e = 0; e < 16; ++e) {
            int k = kc * 32 + 2 * kb + e;
            a[e] = (_Float16)Af[wave * 16 + m][k];
        }
        #pragma unroll
        for (int ct = 0; ct < 4; ++ct) {
            v16h bf = Wf[(kc * 4 + ct) * 32 + lane];
            acc[ct] = __builtin_amdgcn_wmma_f32_16x16x32_f16(
                false, a, false, bf, (short)0, acc[ct], false, false);
        }
    }
    #pragma unroll
    for (int ct = 0; ct < 4; ++ct) {
        int col = ct * 16 + (lane & 15);
        float bv = bias[col];
        #pragma unroll
        for (int r = 0; r < 8; ++r) {
            long row = rowBase + wave * 16 + r + ((lane >> 4) * 8);
            float v = acc[ct][r] + bv;
            xout[row * Dd + col] = v > 0.f ? v : 0.f;
        }
    }
}

// gated init for s>0:  g = sigmoid(x @ gate_W + gate_b);  new = deg*poi*g + (1-g)*x
__global__ __launch_bounds__(128) void pet_gate_gemm_kernel(
        const float* __restrict__ xin, const _Float16* __restrict__ Wswz,
        const float* __restrict__ gate_b, const float* __restrict__ deg,
        const float* __restrict__ poi, float* __restrict__ initout,
        float* __restrict__ xout) {
    __shared__ __align__(16) float Af[64][64];
    const int t = threadIdx.x;
    const int wave = t >> 5, lane = t & 31;
    const long rowBase = (long)blockIdx.x * 64;

    { // stage A tile (64x64 f32) via async DMA
        int r = t >> 1, half = t & 1;
        const float* srcp = xin + (rowBase + r) * Dd + half * 32;
        #pragma unroll
        for (int c4 = 0; c4 < 8; ++c4)
            pet_copy_b128_to_lds(srcp + c4 * 4, &Af[r][half * 32 + c4 * 4]);
    }
    pet_async_wait();
    __syncthreads();

    const v16h* Wf = (const v16h*)Wswz;
    v8f acc[4] = {};
    const int m = lane & 15;
    const int kb = (lane >> 4) * 8;
    #pragma unroll
    for (int kc = 0; kc < 2; ++kc) {
        v16h a;
        #pragma unroll
        for (int e = 0; e < 16; ++e) {
            int k = kc * 32 + 2 * kb + e;
            a[e] = (_Float16)Af[wave * 16 + m][k];
        }
        #pragma unroll
        for (int ct = 0; ct < 4; ++ct) {
            v16h bf = Wf[(kc * 4 + ct) * 32 + lane];
            acc[ct] = __builtin_amdgcn_wmma_f32_16x16x32_f16(
                false, a, false, bf, (short)0, acc[ct], false, false);
        }
    }
    #pragma unroll
    for (int ct = 0; ct < 4; ++ct) {
        int col = ct * 16 + (lane & 15);
        float bv = gate_b[col];
        #pragma unroll
        for (int r = 0; r < 8; ++r) {
            long row = rowBase + wave * 16 + r + ((lane >> 4) * 8);
            int n = (int)(row % Nn);
            float g = 1.f / (1.f + __expf(-(acc[ct][r] + bv)));
            float iv = deg[n] * poi[(long)n * Dd + col];
            float xold = xin[row * Dd + col];
            float nv = iv * g + (1.f - g) * xold;
            initout[row * Dd + col] = nv;
            xout[row * Dd + col]    = nv;
        }
    }
}

// query = (query + g_time_emb[g_time[s]]) @ g_lin_W + g_lin_b    (single block, 512 threads)
__global__ __launch_bounds__(512) void pet_query_kernel(
        const float* __restrict__ g_time_emb, const int* __restrict__ g_time, int s,
        const float* __restrict__ W, const float* __restrict__ bias,
        float* __restrict__ query) {
    __shared__ float qin[Bb][Dd];
    int t = threadIdx.x; int b = t >> 6, d = t & 63;
    qin[b][d] = query[b * Dd + d] + g_time_emb[g_time[s] * Dd + d];
    __syncthreads();
    float acc = bias[d];
    for (int k = 0; k < Dd; ++k) acc += qin[b][k] * W[k * Dd + d];
    query[b * Dd + d] = acc;
}

// xseq[b,h,:] = concat(feature[b,poi,:], query[b,:]) + history_time_encoding
__global__ void pet_gather_hist_kernel(const float* __restrict__ x, const float* __restrict__ query,
                                       const int* __restrict__ hpoi, const float* __restrict__ hte,
                                       float* __restrict__ xseq) {
    long id = (long)blockIdx.x * blockDim.x + threadIdx.x;   // B*HL*128
    if (id >= (long)Bb * HLn * D2d) return;
    int c = id & 127; long bh = id >> 7;
    int h = (int)(bh % HLn); int b = (int)(bh / HLn);
    int p = hpoi[b * HLn + h];
    float v = (c < Dd) ? x[((long)b * Nn + p) * Dd + c] : query[b * Dd + (c - Dd)];
    xseq[id] = v + hte[id];
}

// QKV = xseq @ Wqkv[l]
__global__ void pet_qkv_kernel(const float* __restrict__ xseq, const float* __restrict__ Wqkv,
                               float* __restrict__ qkv) {
    long id = (long)blockIdx.x * blockDim.x + threadIdx.x;   // B*HL*384
    if (id >= (long)Bb * HLn * 384) return;
    int c = (int)(id % 384); long bh = id / 384;
    const float* xr = xseq + bh * D2d;
    float acc = 0.f;
    for (int k = 0; k < D2d; ++k) acc += xr[k] * Wqkv[(long)k * 384 + c];
    qkv[id] = acc;
}

// one attention row (b, i, head): block of 64 threads
__global__ __launch_bounds__(64) void pet_attn_kernel(const float* __restrict__ qkv,
                                                      float* __restrict__ ao) {
    int blk = blockIdx.x;                 // (b*50 + i)*2 + h
    int h = blk & 1; int bi = blk >> 1;
    int i = bi % HLn; int b = bi / HLn;
    __shared__ float qv[64]; __shared__ float sc[HLn]; __shared__ float sinv;
    int t = threadIdx.x;
    const float* base = qkv + (long)b * HLn * 384;
    qv[t] = base[(long)i * 384 + h * 64 + t];
    __syncthreads();
    if (t < HLn) {
        const float* kr = base + (long)t * 384 + 128 + h * 64;
        float s = 0.f;
        for (int d = 0; d < 64; ++d) s += qv[d] * kr[d];
        sc[t] = s * 0.125f;               // 1/sqrt(64)
    }
    __syncthreads();
    if (t == 0) {
        float m = -1e30f;
        for (int j = 0; j < HLn; ++j) m = fmaxf(m, sc[j]);
        float sum = 0.f;
        for (int j = 0; j < HLn; ++j) { float e = __expf(sc[j] - m); sc[j] = e; sum += e; }
        sinv = 1.f / sum;
    }
    __syncthreads();
    float o = 0.f;
    for (int j = 0; j < HLn; ++j) o += sc[j] * base[(long)j * 384 + 256 + h * 64 + t];
    ao[((long)b * HLn + i) * D2d + h * 64 + t] = o * sinv;
}

// xseq = LN(xseq + ao @ Wo)  (one row per block, 128 threads)
__global__ __launch_bounds__(128) void pet_proj_ln_kernel(
        const float* __restrict__ ao, const float* __restrict__ Wo,
        const float* __restrict__ ln, float* __restrict__ xseq) {
    int row = blockIdx.x; int c = threadIdx.x;
    __shared__ float a[D2d]; __shared__ float red[D2d];
    a[c] = ao[(long)row * D2d + c];
    __syncthreads();
    float acc = xseq[(long)row * D2d + c];
    for (int k = 0; k < D2d; ++k) acc += a[k] * Wo[(long)k * D2d + c];
    red[c] = acc; __syncthreads();
    for (int o = 64; o > 0; o >>= 1) { if (c < o) red[c] += red[c + o]; __syncthreads(); }
    float mean = red[0] * (1.f / 128.f); __syncthreads();
    float dv = acc - mean;
    red[c] = dv * dv; __syncthreads();
    for (int o = 64; o > 0; o >>= 1) { if (c < o) red[c] += red[c + o]; __syncthreads(); }
    float var = red[0] * (1.f / 128.f);
    xseq[(long)row * D2d + c] = dv * rsqrtf(var + 1e-5f) * ln[c] + ln[D2d + c];
}

// xseq = LN(xseq + relu(xseq@W1+b1)@W2 + b2)   (one row per block, 128 threads)
__global__ __launch_bounds__(128) void pet_ffn_kernel(
        const float* __restrict__ W1, const float* __restrict__ b1,
        const float* __restrict__ W2, const float* __restrict__ b2,
        const float* __restrict__ ln, float* __restrict__ xseq) {
    int row = blockIdx.x; int t = threadIdx.x;
    __shared__ float xr[D2d]; __shared__ float h1[4 * D2d]; __shared__ float red[D2d];
    xr[t] = xseq[(long)row * D2d + t];
    __syncthreads();
    for (int c = t; c < 4 * D2d; c += 128) {
        float acc = b1[c];
        for (int k = 0; k < D2d; ++k) acc += xr[k] * W1[(long)k * 4 * D2d + c];
        h1[c] = acc > 0.f ? acc : 0.f;
    }
    __syncthreads();
    float acc = b2[t] + xr[t];
    for (int k = 0; k < 4 * D2d; ++k) acc += h1[k] * W2[(long)k * D2d + t];
    red[t] = acc; __syncthreads();
    for (int o = 64; o > 0; o >>= 1) { if (t < o) red[t] += red[t + o]; __syncthreads(); }
    float mean = red[0] * (1.f / 128.f); __syncthreads();
    float dv = acc - mean;
    red[t] = dv * dv; __syncthreads();
    for (int o = 64; o > 0; o >>= 1) { if (t < o) red[t] += red[t + o]; __syncthreads(); }
    float var = red[0] * (1.f / 128.f);
    xseq[(long)row * D2d + t] = dv * rsqrtf(var + 1e-5f) * ln[t] + ln[D2d + t];
}

__global__ void pet_label_mean_kernel(const float* __restrict__ xseq, float* __restrict__ flab) {
    int id = blockIdx.x * blockDim.x + threadIdx.x;   // B*128
    if (id >= Bb * D2d) return;
    int c = id & 127; int b = id >> 7;
    float s = 0.f;
    for (int h = 0; h < HLn; ++h) s += xseq[((long)b * HLn + h) * D2d + c];
    flab[id] = s * (1.f / HLn);
}

// score: ft = [feature[b,t,:]|flab[b,:]] (192) -> relu(ft@W1+b1) -> @W2+b2
__global__ __launch_bounds__(192) void pet_score_kernel(
        const float* __restrict__ x, const float* __restrict__ flab,
        const int* __restrict__ t_index,
        const float* __restrict__ W1, const float* __restrict__ b1,
        const float* __restrict__ W2, const float* __restrict__ b2,
        float* __restrict__ out) {
    int bk = blockIdx.x;                  // B*K
    int b = bk / Kk;
    int t = threadIdx.x;
    __shared__ float ft[192]; __shared__ float red[192];
    int tgt = t_index[bk];
    ft[t] = (t < Dd) ? x[((long)b * Nn + tgt) * Dd + t] : flab[b * D2d + (t - Dd)];
    __syncthreads();
    float acc = b1[t];
    for (int k = 0; k < 192; ++k) acc += ft[k] * W1[(long)k * 192 + t];
    acc = acc > 0.f ? acc : 0.f;
    red[t] = acc * W2[t];
    __syncthreads();
    for (int o = 96; o >= 3; o >>= 1) { if (t < o) red[t] += red[t + o]; __syncthreads(); }
    if (t == 0) out[bk] = red[0] + red[1] + red[2] + b2[0];
}

// ---------------- host orchestration ----------------
extern "C" void kernel_launch(void* const* d_in, const int* in_sizes, int n_in,
                              void* d_out, int out_size, void* d_ws, size_t ws_size,
                              hipStream_t stream) {
    const float* poi_emb   = (const float*)d_in[0];
    const float* query_emb = (const float*)d_in[1];
    const float* gate_W    = (const float*)d_in[2];
    const float* gate_b    = (const float*)d_in[3];
    const float* gnn_rel   = (const float*)d_in[4];   // [2,24,64]
    const float* gnn_W     = (const float*)d_in[5];   // [2,128,64]
    const float* gnn_b     = (const float*)d_in[6];   // [2,64]
    const float* g_time_emb= (const float*)d_in[7];
    const float* g_lin_W   = (const float*)d_in[8];
    const float* g_lin_b   = (const float*)d_in[9];
    const float* tf_Wqkv   = (const float*)d_in[10];  // [2,128,384]
    const float* tf_Wo     = (const float*)d_in[11];  // [2,128,128]
    const float* tf_ln1    = (const float*)d_in[12];  // [2,2,128]
    const float* tf_ln2    = (const float*)d_in[13];
    const float* tf_W1     = (const float*)d_in[14];  // [2,128,512]
    const float* tf_b1     = (const float*)d_in[15];  // [2,512]
    const float* tf_W2     = (const float*)d_in[16];  // [2,512,128]
    const float* tf_b2     = (const float*)d_in[17];  // [2,128]
    const float* mlp_W1    = (const float*)d_in[18];  // [192,192]
    const float* mlp_b1    = (const float*)d_in[19];
    const float* mlp_W2    = (const float*)d_in[20];  // [192,1]
    const float* mlp_b2    = (const float*)d_in[21];
    const float* hte       = (const float*)d_in[22];  // [8,50,128]
    const int* edge_src    = (const int*)d_in[23];    // [3,60000]
    const int* edge_dst    = (const int*)d_in[24];
    const int* edge_type   = (const int*)d_in[25];
    const int* r_index     = (const int*)d_in[27];
    const int* t_index     = (const int*)d_in[28];    // [8,101]
    const int* history_poi = (const int*)d_in[29];    // [8,50]
    const int* g_time      = (const int*)d_in[30];    // [3]

    // workspace layout (floats)
    float* base = (float*)d_ws;
    float* X    = base;                          // 160000*64
    float* INIT = X    + (long)Mrows * Dd;
    float* AGG  = INIT + (long)Mrows * Dd;
    float* DEG  = AGG  + (long)Mrows * Dd;       // 20000
    float* QUERY= DEG  + Nn;                     // 512
    float* XSEQ = QUERY+ Bb * Dd;                // 8*50*128
    float* QKV  = XSEQ + (long)Bb * HLn * D2d;   // 8*50*384
    float* AO   = QKV  + (long)Bb * HLn * 384;   // 8*50*128
    float* FLAB = AO   + (long)Bb * HLn * D2d;   // 8*128
    _Float16* WH      = (_Float16*)(FLAB + Bb * D2d);   // 32B-aligned by construction
    _Float16* WH_GATE = WH;                      // 8 frags * 32 lanes * 16 halfs
    _Float16* WH_GNN  = WH + 64 * 64;            // 2 * 16 frags * 32 * 16

    pet_prep_kernel<<<64, 256, 0, stream>>>(gate_W, gnn_W, query_emb, r_index,
                                            WH_GATE, WH_GNN, QUERY);

    for (int s = 0; s < Ss; ++s) {
        // per-node degree of this snapshot
        pet_zero_kernel<<<20, 256, 0, stream>>>((float4*)DEG, Nn / 4);
        pet_degree_kernel<<<(2 * Ee + 255) / 256, 256, 0, stream>>>(
            edge_src + (long)s * Ee, edge_dst + (long)s * Ee, DEG);

        if (s == 0) {
            pet_init0_kernel<<<(Mrows * 16 + 255) / 256, 256, 0, stream>>>(
                poi_emb, DEG, X, INIT);
        } else {
            pet_gate_gemm_kernel<<<Mrows / 64, 128, 0, stream>>>(
                X, WH_GATE, gate_b, DEG, poi_emb, INIT, X);
        }

        for (int l = 0; l < 2; ++l) {
            pet_zero_kernel<<<(Mrows * 16 + 255) / 256, 256, 0, stream>>>(
                (float4*)AGG, (long)Mrows * 16);
            pet_scatter_kernel<<<((long)Ee * Bb * 16 + 255) / 256, 256, 0, stream>>>(
                edge_src + (long)s * Ee, edge_dst + (long)s * Ee, edge_type + (long)s * Ee,
                X, gnn_rel + (long)l * Rr * Dd, AGG);
            pet_gnn_gemm_kernel<<<Mrows / 64, 128, 0, stream>>>(
                AGG, INIT, WH_GNN + (long)l * 128 * 64, gnn_b + l * Dd, X);
        }
        pet_query_kernel<<<1, 512, 0, stream>>>(g_time_emb, g_time, s, g_lin_W, g_lin_b, QUERY);
    }

    pet_gather_hist_kernel<<<(Bb * HLn * D2d + 255) / 256, 256, 0, stream>>>(
        X, QUERY, history_poi, hte, XSEQ);

    for (int l = 0; l < 2; ++l) {
        pet_qkv_kernel<<<(Bb * HLn * 384 + 255) / 256, 256, 0, stream>>>(
            XSEQ, tf_Wqkv + (long)l * D2d * 384, QKV);
        pet_attn_kernel<<<Bb * HLn * 2, 64, 0, stream>>>(QKV, AO);
        pet_proj_ln_kernel<<<Bb * HLn, 128, 0, stream>>>(
            AO, tf_Wo + (long)l * D2d * D2d, tf_ln1 + (long)l * 2 * D2d, XSEQ);
        pet_ffn_kernel<<<Bb * HLn, 128, 0, stream>>>(
            tf_W1 + (long)l * D2d * 4 * D2d, tf_b1 + (long)l * 4 * D2d,
            tf_W2 + (long)l * 4 * D2d * D2d, tf_b2 + (long)l * D2d,
            tf_ln2 + (long)l * 2 * D2d, XSEQ);
    }

    pet_label_mean_kernel<<<4, 256, 0, stream>>>(XSEQ, FLAB);
    pet_score_kernel<<<Bb * Kk, 192, 0, stream>>>(
        X, FLAB, t_index, mlp_W1, mlp_b1, mlp_W2, mlp_b2, (float*)d_out);
}
